// LSTM_60722247631102
// MI455X (gfx1250) — matrix operand order repaired and checked by reading, not compiled, not run on hardware
//
#include <hip/hip_runtime.h>

// ---------------------------------------------------------------------------
// Types for CDNA5 WMMA (wave32): V_WMMA_F32_16X16X32_BF16
// ---------------------------------------------------------------------------
typedef __bf16 bf16_t;
typedef bf16_t v16bf __attribute__((ext_vector_type(16)));
typedef float  v8f   __attribute__((ext_vector_type(8)));

__device__ __forceinline__ bf16_t to_bf16(float f) {
  unsigned u; __builtin_memcpy(&u, &f, 4);
  u = (u + 0x7FFFu + ((u >> 16) & 1u)) >> 16;   // round-to-nearest-even
  unsigned short s = (unsigned short)u;
  bf16_t r; __builtin_memcpy(&r, &s, 2);
  return r;
}

__device__ __forceinline__ float sigm(float x) { return 1.0f / (1.0f + expf(-x)); }

// ---------------------------------------------------------------------------
// Pack fp32 activations X (logical rows r = t*B + b, source layout [B,T,K])
// into bf16 WMMA A-fragment order (ISA 7.12.2, 16-bit A 16x32):
//   tile (mblk,kblk): lane l (half=l>>4, row=mblk*16+(l&15)), elem e:
//     e<8  -> K = kblk*32 + half*8 + e
//     e>=8 -> K = kblk*32 + 16 + half*8 + (e-8)
// Packed index = ((mblk*K_tiles + kblk)*512) + l*16 + e   (contiguous per lane)
// ---------------------------------------------------------------------------
__global__ void pack_X_kernel(const float* __restrict__ X, bf16_t* __restrict__ out,
                              int Bb, int T, int K) {
  long long idx = (long long)blockIdx.x * blockDim.x + threadIdx.x;
  long long total = (long long)Bb * T * K;
  if (idx >= total) return;
  int e = (int)(idx & 15);
  long long r5 = idx >> 4;
  int lane = (int)(r5 & 31);
  long long tile = r5 >> 5;
  int K_tiles = K >> 5;
  int kblk = (int)(tile % K_tiles);
  long long mblk = tile / K_tiles;
  int m = (int)(mblk * 16 + (lane & 15));          // global packed row r = t*Bb + b
  int half = lane >> 4;
  int kk = (e < 8) ? (half * 8 + e) : (16 + half * 8 + (e - 8));
  int k = kblk * 32 + kk;
  int b = m % Bb;
  int t = m / Bb;
  out[idx] = to_bf16(X[((size_t)b * T + t) * K + k]);
}

// ---------------------------------------------------------------------------
// Pack fp32 weight W [N,K] (row-major) into bf16 WMMA B-fragment order
// (B is K x N, lane = column):
//   tile (nblk,kblk): lane l -> col n = nblk*16 + (l&15); elem e -> K = kblk*32 + (e<8 ? (l>>4)*8 + e : 16 + (l>>4)*8 + (e-8))
// ---------------------------------------------------------------------------
__global__ void pack_W_kernel(const float* __restrict__ W, bf16_t* __restrict__ out,
                              int N, int K) {
  long long idx = (long long)blockIdx.x * blockDim.x + threadIdx.x;
  long long total = (long long)N * K;
  if (idx >= total) return;
  int e = (int)(idx & 15);
  long long r5 = idx >> 4;
  int lane = (int)(r5 & 31);
  long long tile = r5 >> 5;
  int K_tiles = K >> 5;
  int kblk = (int)(tile % K_tiles);
  long long nblk = tile / K_tiles;
  int n = (int)(nblk * 16 + (lane & 15));
  int k = kblk * 32 + ((e < 8) ? ((lane >> 4) * 8 + e) : (16 + (lane >> 4) * 8 + (e - 8)));
  out[idx] = to_bf16(W[(size_t)n * K + k]);
}

__global__ void add_bias_kernel(const float* __restrict__ a, const float* __restrict__ b,
                                float* __restrict__ out, int n) {
  int i = blockIdx.x * blockDim.x + threadIdx.x;
  if (i < n) out[i] = a[i] + b[i];
}

// ---------------------------------------------------------------------------
// WMMA GEMM with 4x4 register blocking + ping-pong double buffering.
// One wave computes a 64x64 output strip: 16 v8f accumulators; per k-step
// 8 fragment loads feed 16 back-to-back WMMAs. K_tiles must be even.
// out[row, col] = sum_k A[row,k]*W[col,k] (+ bias[col]) (+ addsrc[row,col])
// out_mode 0: row-major [row, ldc]; out_mode 1: row=t*Bdim+b -> out[b, t, ldc]
// ---------------------------------------------------------------------------
__global__ __launch_bounds__(256) void wmma_gemm_bf16(
    const bf16_t* __restrict__ Apk, const bf16_t* __restrict__ Bpk,
    float* __restrict__ outp, const float* __restrict__ addsrc,
    const float* __restrict__ bias,
    int M_tiles, int N_tiles, int K_tiles, int ldc,
    int out_mode, int Bdim, int Tdim) {
  const int lane = threadIdx.x & 31;
  const int wave = blockIdx.x * (blockDim.x >> 5) + (threadIdx.x >> 5);
  const int mgroups = M_tiles >> 2;
  const int ngroups = N_tiles >> 2;
  if (wave >= mgroups * ngroups) return;   // wave-uniform exit: EXEC all-ones at WMMA
  const int mg = wave / ngroups;
  const int ng = wave - mg * ngroups;

  const size_t tstride = (size_t)K_tiles * 512;   // elems per tile-row of k-tiles
  const bf16_t* abase = Apk + (size_t)(mg << 2) * tstride + (size_t)lane * 16;
  const bf16_t* bbase = Bpk + (size_t)(ng << 2) * tstride + (size_t)lane * 16;

  v8f acc[16];
#pragma unroll
  for (int i = 0; i < 16; ++i) acc[i] = v8f{};

  v16bf a0[4], b0[4], a1[4], b1[4];

  auto load_frags = [&](v16bf (&a)[4], v16bf (&b)[4], int kk) {
    const size_t off = (size_t)kk * 512;
#pragma unroll
    for (int i = 0; i < 4; ++i) a[i] = *(const v16bf*)(abase + (size_t)i * tstride + off);
#pragma unroll
    for (int j = 0; j < 4; ++j) b[j] = *(const v16bf*)(bbase + (size_t)j * tstride + off);
  };
  auto mma16 = [&](const v16bf (&a)[4], const v16bf (&b)[4]) {
#pragma unroll
    for (int i = 0; i < 4; ++i)
#pragma unroll
      for (int j = 0; j < 4; ++j)
        acc[i * 4 + j] = __builtin_amdgcn_wmma_f32_16x16x32_bf16(
            false, a[i], false, b[j], (short)0, acc[i * 4 + j], false, false);
  };

  load_frags(a0, b0, 0);
  for (int k = 0; k < K_tiles; k += 2) {   // K_tiles even: no remainder
    load_frags(a1, b1, k + 1);
    __builtin_prefetch(abase + (size_t)(k + 3) * 512, 0, 3);
    mma16(a0, b0);
    if (k + 2 < K_tiles) load_frags(a0, b0, k + 2);
    mma16(a1, b1);
  }

  // C/D layout: VGPR v, lane half -> row m = v + 8*half; col = lane&15
  const int half = lane >> 4;
  const int cl = lane & 15;
#pragma unroll
  for (int i = 0; i < 4; ++i) {
    const int rowbase = ((mg << 2) + i) * 16 + half * 8;
#pragma unroll
    for (int j = 0; j < 4; ++j) {
      const int col = ((ng << 2) + j) * 16 + cl;
      const float badd = bias ? bias[col] : 0.0f;
#pragma unroll
      for (int v = 0; v < 8; ++v) {
        const int row = rowbase + v;
        float val = acc[i * 4 + j][v] + badd;
        if (addsrc) val += addsrc[(size_t)row * ldc + col];
        if (out_mode == 0) {
          outp[(size_t)row * ldc + col] = val;
        } else {
          const int bb = row % Bdim;
          const int tt = row / Bdim;
          outp[((size_t)bb * Tdim + tt) * ldc + col] = val;
        }
      }
    }
  }
}

// ---------------------------------------------------------------------------
// LSTM pointwise gate update. gates: [Bb, 4H] fp32 preactivations (i,f,g,o).
// Writes c (fp32 state) and h directly in packed-A bf16 order at rows t*Bb+b.
// ---------------------------------------------------------------------------
__global__ void lstm_pointwise(const float* __restrict__ gates,
                               float* __restrict__ cstate,
                               bf16_t* __restrict__ hpack,
                               int t, int Bb, int H, int is_first) {
  int idx = blockIdx.x * blockDim.x + threadIdx.x;
  if (idx >= Bb * H) return;
  int b = idx / H;
  int h = idx - b * H;
  const float* g = gates + (size_t)b * 4 * H;
  float gi = sigm(g[h]);
  float gf = sigm(g[H + h]);
  float gg = tanhf(g[2 * H + h]);
  float go = sigm(g[3 * H + h]);
  float cprev = is_first ? 0.0f : cstate[idx];
  float cn = gf * cprev + gi * gg;
  cstate[idx] = cn;
  float hn = go * tanhf(cn);

  // inverse of A-fragment mapping for row r = t*Bb + b, K index h
  int r = t * Bb + b;
  int mblk = r >> 4;
  int K_tiles = H >> 5;
  int kblk = h >> 5;
  int kk = h & 31;
  int half, e;
  if (kk < 16) { half = kk >> 3; e = kk & 7; }
  else         { half = (kk - 16) >> 3; e = 8 + ((kk - 16) & 7); }
  int lane = half * 16 + (r & 15);
  size_t dst = ((size_t)mblk * K_tiles + kblk) * 512 + (size_t)lane * 16 + e;
  hpack[dst] = to_bf16(hn);
}

// ---------------------------------------------------------------------------
extern "C" void kernel_launch(void* const* d_in, const int* in_sizes, int n_in,
                              void* d_out, int out_size, void* d_ws, size_t ws_size,
                              hipStream_t stream) {
  (void)in_sizes; (void)n_in; (void)out_size; (void)ws_size;
  const int B = 128, T = 512, I = 512, H = 1024, FH = 4096, O = 512;
  const int ROWS = B * T;  // packed row r = t*B + b

  const float* x     = (const float*)d_in[0];
  const float* W_ih0 = (const float*)d_in[1];
  const float* W_hh0 = (const float*)d_in[2];
  const float* b_ih0 = (const float*)d_in[3];
  const float* b_hh0 = (const float*)d_in[4];
  const float* W_ih1 = (const float*)d_in[5];
  const float* W_hh1 = (const float*)d_in[6];
  const float* b_ih1 = (const float*)d_in[7];
  const float* b_hh1 = (const float*)d_in[8];
  const float* W_lin = (const float*)d_in[9];
  const float* b_lin = (const float*)d_in[10];
  float* out = (float*)d_out;

  // workspace carve-out
  char* wp = (char*)d_ws;
  auto carve = [&](size_t bytes) { void* p = (void*)wp; wp += (bytes + 255) & ~(size_t)255; return p; };
  bf16_t* xpack  = (bf16_t*)carve((size_t)ROWS * I * 2);
  float*  xg     = (float*) carve((size_t)ROWS * FH * 4);
  bf16_t* h1pack = (bf16_t*)carve((size_t)ROWS * H * 2);
  bf16_t* h2pack = (bf16_t*)carve((size_t)ROWS * H * 2);
  bf16_t* wih0p  = (bf16_t*)carve((size_t)FH * I * 2);
  bf16_t* whh0p  = (bf16_t*)carve((size_t)FH * H * 2);
  bf16_t* wih1p  = (bf16_t*)carve((size_t)FH * H * 2);
  bf16_t* whh1p  = (bf16_t*)carve((size_t)FH * H * 2);
  bf16_t* wlinp  = (bf16_t*)carve((size_t)O * H * 2);
  float*  bias0  = (float*) carve((size_t)FH * 4);
  float*  bias1  = (float*) carve((size_t)FH * 4);
  float*  cstate = (float*) carve((size_t)B * H * 4);
  float*  gates  = (float*) carve((size_t)B * FH * 4);

  auto blocks_for = [](long long n) { return (unsigned)((n + 255) / 256); };

  // --- pack weights & activations to bf16 fragment order ---
  pack_X_kernel<<<blocks_for((long long)ROWS * I), 256, 0, stream>>>(x, xpack, B, T, I);
  pack_W_kernel<<<blocks_for((long long)FH * I), 256, 0, stream>>>(W_ih0, wih0p, FH, I);
  pack_W_kernel<<<blocks_for((long long)FH * H), 256, 0, stream>>>(W_hh0, whh0p, FH, H);
  pack_W_kernel<<<blocks_for((long long)FH * H), 256, 0, stream>>>(W_ih1, wih1p, FH, H);
  pack_W_kernel<<<blocks_for((long long)FH * H), 256, 0, stream>>>(W_hh1, whh1p, FH, H);
  pack_W_kernel<<<blocks_for((long long)O * H), 256, 0, stream>>>(W_lin, wlinp, O, H);
  add_bias_kernel<<<blocks_for(FH), 256, 0, stream>>>(b_ih0, b_hh0, bias0, FH);
  add_bias_kernel<<<blocks_for(FH), 256, 0, stream>>>(b_ih1, b_hh1, bias1, FH);

  auto gemm = [&](const bf16_t* A, const bf16_t* Bp, float* op, const float* add,
                  const float* bias, int Mt, int Nt, int Kt, int ldc, int mode) {
    long long waves = (long long)(Mt >> 2) * (Nt >> 2);
    unsigned nblk = (unsigned)((waves * 32 + 255) / 256);
    wmma_gemm_bf16<<<nblk, 256, 0, stream>>>(A, Bp, op, add, bias, Mt, Nt, Kt, ldc, mode, B, T);
  };

  const int htiles_per_step = B / 16;               // 8 mtiles per timestep
  const size_t h_mtile_elems = (size_t)(H / 32) * 512;

  // ================= Layer 0 =================
  gemm(xpack, wih0p, xg, nullptr, bias0, ROWS / 16, FH / 16, I / 32, FH, 0);
  for (int t = 0; t < T; ++t) {
    const float* gsrc;
    if (t > 0) {
      const bf16_t* aprev = h1pack + (size_t)(t - 1) * htiles_per_step * h_mtile_elems;
      gemm(aprev, whh0p, gates, xg + (size_t)t * B * FH, nullptr,
           htiles_per_step, FH / 16, H / 32, FH, 0);
      gsrc = gates;
    } else {
      gsrc = xg;  // h_prev = 0 -> gates are just xg rows for t=0
    }
    lstm_pointwise<<<blocks_for((long long)B * H), 256, 0, stream>>>(
        gsrc, cstate, h1pack, t, B, H, (t == 0) ? 1 : 0);
  }

  // ================= Layer 1 =================
  gemm(h1pack, wih1p, xg, nullptr, bias1, ROWS / 16, FH / 16, H / 32, FH, 0);
  for (int t = 0; t < T; ++t) {
    const float* gsrc;
    if (t > 0) {
      const bf16_t* aprev = h2pack + (size_t)(t - 1) * htiles_per_step * h_mtile_elems;
      gemm(aprev, whh1p, gates, xg + (size_t)t * B * FH, nullptr,
           htiles_per_step, FH / 16, H / 32, FH, 0);
      gsrc = gates;
    } else {
      gsrc = xg;
    }
    lstm_pointwise<<<blocks_for((long long)B * H), 256, 0, stream>>>(
        gsrc, cstate, h2pack, t, B, H, (t == 0) ? 1 : 0);
  }

  // ================= Final linear: out[b,t,o] = h2 @ W_lin^T + b_lin =========
  gemm(h2pack, wlinp, out, nullptr, b_lin, ROWS / 16, O / 16, H / 32, O, 1);
}
